// SimpleMultiHeadAttention_1374389535053
// MI455X (gfx1250) — matrix operand order repaired and checked by reading, not compiled, not run on hardware
//
#include <hip/hip_runtime.h>
#include <hip/hip_bf16.h>

// ---------------------------------------------------------------------------
// Fused single-head attention with QKV projections for MI455X (gfx1250).
//   q = query @ Wq^T + bq ; k = key @ Wk^T + bk ; v = value @ Wv^T + bv
//   out = softmax(mask(q k^T / sqrt(E))) v
// bf16 WMMA (v_wmma_f32_16x16x32_bf16) for all GEMMs, flash-style online
// softmax (parallelized across all 256 threads) so the 4096x4096 score
// matrix never touches HBM.
// Workspace layout (d_ws, needs 96 MB):
//   [0)        q  bf16 [B*L, E]
//   [32 MB)    k  bf16 [B*S, E]   (row-major: QK^T B-operand reads 16 contig e)
//   [64 MB)    vT bf16 [B, E, S]  (transposed: PV B-operand reads 16 contig s)
// ---------------------------------------------------------------------------

typedef __attribute__((ext_vector_type(16))) __bf16 v16bf;
typedef __attribute__((ext_vector_type(8)))  float  v8f;

#define BATCH 4
#define LQ    4096
#define SKDIM 4096
#define EDIM  1024
#define STILE 256
#define NEG_BIG (-1.0e30f)

__device__ __forceinline__ v8f wmma_bf16(v16bf a, v16bf b, v8f c) {
  // D = A(16x32) * B(32x16) + C(16x16 f32)
  return __builtin_amdgcn_wmma_f32_16x16x32_bf16(
      /*neg_a=*/false, a, /*neg_b=*/false, b,
      /*c_mod=*/(short)0, c, /*reuse_a=*/false, /*reuse_b=*/false);
}

// CDNA5 16-bit A-matrix (16x32) per-lane layout (ISA 7.12.2):
//   lanes 0-15  hold row lr, K = {0..7, 16..23}
//   lanes 16-31 hold row lr, K = {8..15, 24..31}
// -> staged here in contiguous k instead: lane = lr + (k&16 ? 16 : 0); slot = k & 15
// CDNA5 16-bit B-matrix (32x16) per-lane order used here (contiguous k, not the ISA table order):
//   lanes 0-15  hold col n, K = 0..15 ; lanes 16-31 hold col n, K = 16..31
// -> lane = n + (k&16 ? 16 : 0); slot = k & 15
// Fragments staged so each lane's 16 bf16 are contiguous (one 32B LDS read).

// ---------------------------------------------------------------------------
// Projection GEMM: Out[m,n] = sum_k X[m,k]*W[n,k] + bias[n], M=16384, N=K=1024
// Block: 256 threads (8 waves), 128x128 tile; wave = 64(M) x 32(N) subtile.
// ---------------------------------------------------------------------------
__global__ __launch_bounds__(256) void proj_gemm_bf16(
    const float* __restrict__ X, const float* __restrict__ W,
    const float* __restrict__ bias, __bf16* __restrict__ Out,
    int transposeOut) {
  __shared__ __align__(32) __bf16 As[128 * 32];  // 8 A-fragment tiles
  __shared__ __align__(32) __bf16 Bs[128 * 32];  // 8 B-fragment tiles

  const int tid  = threadIdx.x;
  const int lane = tid & 31;
  const int wave = tid >> 5;
  const int wm   = wave & 1;   // 0..1 -> M offset wm*64
  const int wn   = wave >> 1;  // 0..3 -> N offset wn*32
  const long m0  = (long)blockIdx.x * 128;
  const int  n0  = blockIdx.y * 128;

  const v8f vz = {0.f, 0.f, 0.f, 0.f, 0.f, 0.f, 0.f, 0.f};
  v8f acc[4][2];
#pragma unroll
  for (int i = 0; i < 4; ++i)
#pragma unroll
    for (int j = 0; j < 2; ++j) acc[i][j] = vz;

  for (int k0 = 0; k0 < EDIM; k0 += 32) {
    // Stage A tile (128x32) f32 -> bf16, pre-swizzled into fragment order.
    for (int idx = tid; idx < 128 * 32; idx += 256) {
      int r = idx >> 5, k = idx & 31;
      float v = X[(m0 + r) * (long)EDIM + k0 + k];
      int mt = r >> 4, lr = r & 15;
      int ln  = lr + ((k & 16) ? 16 : 0);
      int pos = k & 15;
      As[((mt * 32 + ln) << 4) + pos] = (__bf16)v;
    }
    // Stage B tile: B[k][n] = W[n0+n][k0+k] (the W^T of the torch Linear).
    for (int idx = tid; idx < 128 * 32; idx += 256) {
      int n = idx >> 5, k = idx & 31;
      float v = W[(long)(n0 + n) * EDIM + k0 + k];
      int nt = n >> 4, nc = n & 15;
      int ln  = nc + ((k & 16) ? 16 : 0);
      int pos = k & 15;
      Bs[((nt * 32 + ln) << 4) + pos] = (__bf16)v;
    }
    __syncthreads();

    v16bf a[4], b[2];
#pragma unroll
    for (int i = 0; i < 4; ++i)
      a[i] = *(const v16bf*)&As[((wm * 4 + i) * 32 + lane) << 4];
#pragma unroll
    for (int j = 0; j < 2; ++j)
      b[j] = *(const v16bf*)&Bs[((wn * 2 + j) * 32 + lane) << 4];
#pragma unroll
    for (int i = 0; i < 4; ++i)
#pragma unroll
      for (int j = 0; j < 2; ++j)
        acc[i][j] = wmma_bf16(a[i], b[j], acc[i][j]);
    __syncthreads();
  }

  // Epilogue: C element (vgpr r, lane) -> row = r + (lane&16 ? 8:0), col = lane&15.
  const int rlow = (lane & 16) ? 8 : 0;
  const int colL = lane & 15;
#pragma unroll
  for (int i = 0; i < 4; ++i) {
#pragma unroll
    for (int j = 0; j < 2; ++j) {
      int n = n0 + (wn * 2 + j) * 16 + colL;
      float bv = bias[n];
#pragma unroll
      for (int r = 0; r < 8; ++r) {
        long row = m0 + (wm * 4 + i) * 16 + r + rlow;
        float val = acc[i][j][r] + bv;
        if (!transposeOut) {
          Out[row * EDIM + n] = (__bf16)val;      // [B*S, E]
        } else {
          long bb = row >> 12;                    // row / 4096
          long s  = row & 4095;
          Out[(bb * EDIM + n) * (long)SKDIM + s] = (__bf16)val;  // [B, E, S]
        }
      }
    }
  }
}

// ---------------------------------------------------------------------------
// Flash attention: one 16-row Q block per workgroup, 8 waves.
// Per 256-wide S-tile:
//   P1: wave w computes scores for s-cols [32w,32w+32) (64 WMMA) -> Sc
//   P2a: 256 threads (row, 16-col chunk) do scale+mask, partial max  -> pm
//   P2b: 256 threads combine max, exp, partial sum, P -> A-frag LDS  -> ps
//   P2c: 16 threads fold running stats (m, l, alpha)
//   P3: every wave rescales its private 16x128 f32 accumulator by alpha
//       and does O += P.V (64 WMMA) against transposed V.
// Barrier ordering (4 per tile) proven race-free across iterations:
//   next iter's Sc writes ordered by this iter's sync4; next iter's Pf
//   writes ordered by next iter's sync1/sync2; alph reads (P3) ordered
//   against next write by next iter's sync3.
// ---------------------------------------------------------------------------
__global__ __launch_bounds__(256) void flash_attn_bf16(
    const __bf16* __restrict__ Q, const __bf16* __restrict__ K,
    const __bf16* __restrict__ VT, const unsigned char* __restrict__ mask,
    float* __restrict__ Out) {
  __shared__ __align__(32) __bf16 Qs[16 * EDIM];   // Q in A-frag layout, 32 KB
  __shared__ __align__(32) float  Sc[16 * STILE];  // scores, 16 KB
  __shared__ __align__(32) __bf16 Pf[16 * STILE];  // P in A-frag layout, 8 KB
  __shared__ float pm[256], ps[256];               // partial max / sum, 2 KB
  __shared__ float mrun[16], lrun[16], alph[16];

  const int tid = threadIdx.x, lane = tid & 31, wave = tid >> 5;
  const int m0  = blockIdx.x * 16;
  const int bb  = blockIdx.y;

  const __bf16* Qb = Q  + ((long)bb * LQ + m0) * EDIM;
  const __bf16* Kb = K  + (long)bb * SKDIM * EDIM;
  const __bf16* Vb = VT + (long)bb * EDIM * SKDIM;

  // Stage the 16xE Q block once, pre-swizzled per 32-wide K chunk.
  for (int idx = tid; idx < 16 * EDIM; idx += 256) {
    int lr = idx >> 10, e = idx & (EDIM - 1);
    int kc = e >> 5, k = e & 31;
    int ln  = lr + ((k & 16) ? 16 : 0);
    int pos = k & 15;
    Qs[((kc * 32 + ln) << 4) + pos] = Qb[(long)lr * EDIM + e];
  }
  if (tid < 16) { mrun[tid] = NEG_BIG; lrun[tid] = 0.f; }

  const v8f vz = {0.f, 0.f, 0.f, 0.f, 0.f, 0.f, 0.f, 0.f};
  v8f oacc[8];
#pragma unroll
  for (int t = 0; t < 8; ++t) oacc[t] = vz;
  __syncthreads();

  const int rlow = (lane & 16) ? 8 : 0;
  const int colL = lane & 15;
  const int ksel = (lane & 16) ? 16 : 0;
  const int srow = tid >> 4;   // softmax row owned by this thread
  const int sch  = tid & 15;   // 16-column chunk within the row

  for (int s0 = 0; s0 < SKDIM; s0 += STILE) {
    if (s0 + STILE < SKDIM)  // pull next tile's K rows toward L2/L0
      __builtin_prefetch(Kb + (long)(s0 + STILE + wave * 32 + colL) * EDIM, 0, 1);

    // --- P1: scores S[m, sw..sw+32) = Q . K^T (64 WMMA per wave) ---
    v8f sacc[2];
    sacc[0] = vz; sacc[1] = vz;
    const int sw = s0 + wave * 32;
    for (int kc = 0; kc < EDIM / 32; ++kc) {
      v16bf a = *(const v16bf*)&Qs[(kc * 32 + lane) << 4];
#pragma unroll
      for (int j = 0; j < 2; ++j) {
        // B[e,s] = K[s,e]: lane holds col s = sw+j*16+colL, 16 contiguous e.
        const __bf16* kp = Kb + (long)(sw + j * 16 + colL) * EDIM + kc * 32 + ksel;
        v16bf bf = *(const v16bf*)kp;
        sacc[j] = wmma_bf16(a, bf, sacc[j]);
      }
    }
#pragma unroll
    for (int j = 0; j < 2; ++j)
#pragma unroll
      for (int r = 0; r < 8; ++r)
        Sc[(r + rlow) * STILE + wave * 32 + j * 16 + colL] = sacc[j][r];
    __syncthreads();  // sync1: scores visible; prior-iter Pf reads done

    // --- P2a: scale + mask + partial row max (all 256 threads) ---
    {
      const uint4 m4 = *(const uint4*)(mask + (long)(m0 + srow) * SKDIM + s0 + sch * 16);
      unsigned int mw[4] = {m4.x, m4.y, m4.z, m4.w};
      float lm = NEG_BIG;
#pragma unroll
      for (int jj = 0; jj < 16; ++jj) {
        int col = sch * 16 + jj;
        float x = Sc[srow * STILE + col] * 0.03125f;  // 1/sqrt(1024)
        bool keep = (mw[jj >> 2] >> ((jj & 3) * 8)) & 0xff;
        x = keep ? x : NEG_BIG;
        Sc[srow * STILE + col] = x;
        lm = fmaxf(lm, x);
      }
      pm[tid] = lm;
    }
    __syncthreads();  // sync2: partial maxima visible

    // --- P2b: full row max (replicated), exp, P frags, partial sum ---
    {
      float mcur = mrun[srow];
#pragma unroll
      for (int c = 0; c < 16; ++c) mcur = fmaxf(mcur, pm[srow * 16 + c]);
      float ls = 0.f;
#pragma unroll
      for (int jj = 0; jj < 16; ++jj) {
        int col = sch * 16 + jj;
        float p = __expf(Sc[srow * STILE + col] - mcur);
        ls += p;
        int kc = col >> 5, k = col & 31;
        int ln  = srow + ((k & 16) ? 16 : 0);
        int pos = k & 15;
        Pf[((kc * 32 + ln) << 4) + pos] = (__bf16)p;  // P in A-frag layout
      }
      ps[tid] = ls;
    }
    __syncthreads();  // sync3: P frags + partial sums visible

    // --- P2c: fold running stats (16 threads) ---
    if (tid < 16) {
      float mprev = mrun[tid], mcur = mprev;
#pragma unroll
      for (int c = 0; c < 16; ++c) mcur = fmaxf(mcur, pm[tid * 16 + c]);
      float alpha = __expf(mprev - mcur);  // finite sentinel: no inf-inf NaN
      float l = lrun[tid] * alpha;
#pragma unroll
      for (int c = 0; c < 16; ++c) l += ps[tid * 16 + c];
      mrun[tid] = mcur; lrun[tid] = l; alph[tid] = alpha;
    }
    __syncthreads();  // sync4: alpha/l updated before P3 + epilogue

    // --- P3: rescale O by alpha, then O += P . V (64 WMMA per wave) ---
    float arow[8];
#pragma unroll
    for (int r = 0; r < 8; ++r) arow[r] = alph[r + rlow];
#pragma unroll
    for (int t = 0; t < 8; ++t)
#pragma unroll
      for (int r = 0; r < 8; ++r) oacc[t][r] *= arow[r];

    for (int kc = 0; kc < STILE / 32; ++kc) {
      v16bf a = *(const v16bf*)&Pf[(kc * 32 + lane) << 4];
#pragma unroll
      for (int t = 0; t < 8; ++t) {
        // B[s,e] = V[s,e] from vT[e,s]: lane holds col e, 16 contiguous s.
        int e = wave * 128 + t * 16 + colL;
        const __bf16* vp = Vb + (long)e * SKDIM + s0 + kc * 32 + ksel;
        v16bf bf = *(const v16bf*)vp;
        oacc[t] = wmma_bf16(a, bf, oacc[t]);
      }
    }
  }

  // Epilogue: out = O / l  (f32, [B, L, E]); lrun valid past sync4
#pragma unroll
  for (int t = 0; t < 8; ++t) {
    int e = wave * 128 + t * 16 + colL;
#pragma unroll
    for (int r = 0; r < 8; ++r) {
      int row = r + rlow;
      float l = lrun[row];
      float val = (l > 0.f) ? oacc[t][r] * (1.0f / l) : 0.f;
      Out[((long)bb * LQ + m0 + row) * EDIM + e] = val;
    }
  }
}

// ---------------------------------------------------------------------------
// Launch: 3 projection GEMMs (V transposed) + flash attention.
// ---------------------------------------------------------------------------
extern "C" void kernel_launch(void* const* d_in, const int* in_sizes, int n_in,
                              void* d_out, int out_size, void* d_ws,
                              size_t ws_size, hipStream_t stream) {
  const float* query = (const float*)d_in[0];
  const float* key   = (const float*)d_in[1];
  const float* value = (const float*)d_in[2];
  const unsigned char* attn_mask = (const unsigned char*)d_in[3];  // jax bool = 1B
  const float* Wq = (const float*)d_in[4];
  const float* bq = (const float*)d_in[5];
  const float* Wk = (const float*)d_in[6];
  const float* bk = (const float*)d_in[7];
  const float* Wv = (const float*)d_in[8];
  const float* bv = (const float*)d_in[9];
  float* out = (float*)d_out;

  const size_t elems = (size_t)BATCH * LQ * EDIM;  // 16M elements, 32 MB bf16
  __bf16* qws  = (__bf16*)d_ws;
  __bf16* kws  = qws + elems;
  __bf16* vtws = kws + elems;  // requires ws_size >= 96 MB

  dim3 gproj((BATCH * LQ) / 128, EDIM / 128);  // 128 x 8 blocks
  proj_gemm_bf16<<<gproj, 256, 0, stream>>>(query, Wq, bq, qws, 0);
  proj_gemm_bf16<<<gproj, 256, 0, stream>>>(key, Wk, bk, kws, 0);
  proj_gemm_bf16<<<gproj, 256, 0, stream>>>(value, Wv, bv, vtws, 1);

  dim3 gattn(LQ / 16, BATCH);  // 256 x 4 blocks, 8 waves each
  flash_attn_bf16<<<gattn, 256, 0, stream>>>(qws, kws, vtws, attn_mask, out);
}
